// HydraWaaaghBlock_19258633356053
// MI455X (gfx1250) — compile-verified, matrix operand order checked by reading, not run on hardware
//
#include <hip/hip_runtime.h>

// ---------------------------------------------------------------------------
// CDNA5 (gfx1250, wave32) types
// ---------------------------------------------------------------------------
typedef __bf16 bf16;
typedef __attribute__((ext_vector_type(16))) __bf16 bf16x16;
typedef __attribute__((ext_vector_type(8)))  float  f32x8;
typedef int v4i __attribute__((vector_size(16)));   // int4, matches builtin param

struct __align__(16) bf8v { bf16 v[8]; };   // one 16B chunk (8 bf16)

#define TM 128
#define TN 128
#define TK 32
#define ASTR 40   // LDS row stride (bf16) for A tile; 80B keeps 16B chunk alignment
#define BSTR 40   // LDS row stride for B tile, stored [n][k]

// Async global->LDS (CDNA5 ASYNCcnt path), guarded so the file always builds.
#if defined(__has_builtin)
#  if __has_builtin(__builtin_amdgcn_global_load_async_to_lds_b128) && \
      __has_builtin(__builtin_amdgcn_s_wait_asynccnt)
#    define USE_ASYNC_LDS 1
#  endif
#endif
#ifndef USE_ASYNC_LDS
#  define USE_ASYNC_LDS 0
#endif

// builtin arg0: generic v4i*  (per hipcc diagnostic), arg1: LDS pointer
#define GPV(p)  ((v4i*)(p))
#define LDSV(p) ((__attribute__((address_space(3))) v4i*)(p))

__device__ __forceinline__ float gelu_exact(float x) {
  return 0.5f * x * (1.0f + erff(x * 0.70710678118654752f));
}

// ---------------------------------------------------------------------------
// bf16 WMMA GEMM, single B layout:  C = act(alpha * A@B^T + bias)
//   A  : M x K row-major (batch stride sAb), lda  (lda % 8 == 0)
//   B  : N x K row-major (batch stride sBb), ldb  (ldb % 8 == 0)  == op-B transposed
//   Cf : optional f32 out, Cb: optional bf16 out (ldc), CbT: optional bf16
//        transposed out (ldct, batch stride sCbT)
// Requirements: M % 128 == 0, K % 32 == 0; N guarded.
// Pipeline: double-buffered LDS tiles filled by async global->LDS B128 copies,
// each wave issues exactly 4 async ops per tile so s_wait_asynccnt(4) retires
// the previous tile while the next streams in under the WMMAs.
// ---------------------------------------------------------------------------
__global__ __launch_bounds__(256) void gemm_bt_wmma(
    const bf16* __restrict__ A, long long sAb, int lda,
    const bf16* __restrict__ B, long long sBb, int ldb,
    float* __restrict__ Cf, bf16* __restrict__ Cb,
    bf16* __restrict__ CbT, long long sCb, int ldc, int ldct, long long sCbT,
    const float* __restrict__ bias, float alpha, int actGelu,
    int M, int N, int K)
{
#if USE_ASYNC_LDS
  __shared__ bf16 At[2][TM * ASTR];
  __shared__ bf16 Bt[2][TN * BSTR];
#else
  __shared__ bf16 At[1][TM * ASTR];
  __shared__ bf16 Bt[1][TN * BSTR];
#endif

  const int z = blockIdx.z;
  A += (long long)z * sAb;
  B += (long long)z * sBb;
  const long long coff  = (long long)z * sCb;
  const long long coffT = (long long)z * sCbT;

  const int bm   = blockIdx.y * TM;
  const int bn   = blockIdx.x * TN;
  const int tid  = (int)threadIdx.x;
  const int lane = tid & 31;
  const int wv   = tid >> 5;
  const int waveM = wv & 3;     // 4 waves x 32 rows
  const int waveN = wv >> 2;    // 2 waves x 64 cols
  const int hh   = lane >> 4;
  const int l16  = lane & 15;

  f32x8 acc[2][4];
  const f32x8 zero8 = {0.f,0.f,0.f,0.f,0.f,0.f,0.f,0.f};
#pragma unroll
  for (int i = 0; i < 2; ++i)
#pragma unroll
    for (int j = 0; j < 4; ++j) acc[i][j] = zero8;

  // chunk decomposition: 128 rows x 4 chunks(16B) = 512 chunks, 2 per thread
  const int cr0 = (tid + 0)   >> 2, cc0 = ((tid + 0)   & 3) << 3;
  const int cr1 = (tid + 256) >> 2, cc1 = ((tid + 256) & 3) << 3;
  // clamp B rows (keeps per-wave async-op count exact; OOB cols never stored)
  const int br0 = (bn + cr0 < N) ? (bn + cr0) : (N - 1);
  const int br1 = (bn + cr1 < N) ? (bn + cr1) : (N - 1);

#if USE_ASYNC_LDS
  auto stage = [&](int buf, int k0) {
    __builtin_amdgcn_global_load_async_to_lds_b128(
        GPV(A + (long long)(bm + cr0) * lda + (k0 + cc0)),
        LDSV(&At[buf][cr0 * ASTR + cc0]), 0, 0);
    __builtin_amdgcn_global_load_async_to_lds_b128(
        GPV(A + (long long)(bm + cr1) * lda + (k0 + cc1)),
        LDSV(&At[buf][cr1 * ASTR + cc1]), 0, 0);
    __builtin_amdgcn_global_load_async_to_lds_b128(
        GPV(B + (long long)br0 * ldb + (k0 + cc0)),
        LDSV(&Bt[buf][cr0 * BSTR + cc0]), 0, 0);
    __builtin_amdgcn_global_load_async_to_lds_b128(
        GPV(B + (long long)br1 * ldb + (k0 + cc1)),
        LDSV(&Bt[buf][cr1 * BSTR + cc1]), 0, 0);
  };
  stage(0, 0);
  int buf = 0;
#endif

  for (int k0 = 0; k0 < K; k0 += TK) {
#if USE_ASYNC_LDS
    const int nxt = buf ^ 1;
    if (k0 + TK < K) {
      stage(nxt, k0 + TK);
      __builtin_amdgcn_s_wait_asynccnt(4);   // previous tile fully in LDS
    } else {
      __builtin_amdgcn_s_wait_asynccnt(0);
    }
    __syncthreads();
    const bf16* Ab = At[buf];
    const bf16* Bb = Bt[buf];
#else
    // fallback: batched register roundtrip (all loads issued before stores)
    bf8v ra0, ra1, rb0, rb1;
    ra0 = *reinterpret_cast<const bf8v*>(A + (long long)(bm + cr0) * lda + (k0 + cc0));
    ra1 = *reinterpret_cast<const bf8v*>(A + (long long)(bm + cr1) * lda + (k0 + cc1));
    rb0 = *reinterpret_cast<const bf8v*>(B + (long long)br0 * ldb + (k0 + cc0));
    rb1 = *reinterpret_cast<const bf8v*>(B + (long long)br1 * ldb + (k0 + cc1));
    if (k0 + TK < K) {
      __builtin_prefetch(A + (long long)(bm + cr0) * lda + (k0 + TK + cc0), 0, 1);
      __builtin_prefetch(B + (long long)br0 * ldb + (k0 + TK + cc1), 0, 1);
    }
    __syncthreads();   // previous tile's fragment reads complete
    *reinterpret_cast<bf8v*>(&At[0][cr0 * ASTR + cc0]) = ra0;
    *reinterpret_cast<bf8v*>(&At[0][cr1 * ASTR + cc1]) = ra1;
    *reinterpret_cast<bf8v*>(&Bt[0][cr0 * BSTR + cc0]) = rb0;
    *reinterpret_cast<bf8v*>(&Bt[0][cr1 * BSTR + cc1]) = rb1;
    __syncthreads();
    const bf16* Ab = At[0];
    const bf16* Bb = Bt[0];
#endif

    // fragments per documented 16-bit operand layout:
    // lane(0..15)=row, elem pair v: k = (v<4 ? 2v : 16+2(v-4)) + half*8
    bf16x16 afr[2];
#pragma unroll
    for (int mf = 0; mf < 2; ++mf) {
      const bf16* p = &Ab[(waveM * 32 + mf * 16 + l16) * ASTR];
#pragma unroll
      for (int v = 0; v < 8; ++v) {
        const int kk = ((v < 4) ? (2 * v) : (16 + 2 * (v - 4))) + hh * 8;
        afr[mf][2 * v]     = p[kk];
        afr[mf][2 * v + 1] = p[kk + 1];
      }
    }
    bf16x16 bfr[4];
#pragma unroll
    for (int nf = 0; nf < 4; ++nf) {
      const bf16* p = &Bb[(waveN * 64 + nf * 16 + l16) * BSTR];
#pragma unroll
      for (int v = 0; v < 8; ++v) {
        const int kk = ((v < 4) ? (2 * v) : (16 + 2 * (v - 4))) + hh * 8;
        bfr[nf][2 * v]     = p[kk];
        bfr[nf][2 * v + 1] = p[kk + 1];
      }
    }
#pragma unroll
    for (int mf = 0; mf < 2; ++mf)
#pragma unroll
      for (int nf = 0; nf < 4; ++nf)
        acc[mf][nf] = __builtin_amdgcn_wmma_f32_16x16x32_bf16(
            false, afr[mf], false, bfr[nf], (short)0, acc[mf][nf], false, false);
    __syncthreads();
#if USE_ASYNC_LDS
    buf = nxt;
#endif
  }

  // epilogue: C VGPR r -> row = half*8 + r, col = lane%16
#pragma unroll
  for (int mf = 0; mf < 2; ++mf) {
#pragma unroll
    for (int nf = 0; nf < 4; ++nf) {
      const int col = bn + waveN * 64 + nf * 16 + l16;
      if (col < N) {
#pragma unroll
        for (int r = 0; r < 8; ++r) {
          const int row = bm + waveM * 32 + mf * 16 + hh * 8 + r;
          float vv = acc[mf][nf][r] * alpha;
          if (bias) vv += bias[col];
          if (actGelu) vv = gelu_exact(vv);
          if (Cf)  Cf[coff + (long long)row * ldc + col] = vv;
          if (Cb)  Cb[coff + (long long)row * ldc + col] = (bf16)vv;
          if (CbT) CbT[coffT + (long long)col * ldct + row] = (bf16)vv;
        }
      }
    }
  }
}

// ---------------------------------------------------------------------------
// Tiled transpose + f32->bf16 convert:  src [z][R][C] f32 -> dst [z][C][R] bf16
// ---------------------------------------------------------------------------
__global__ __launch_bounds__(256) void transpose_cvt_kernel(
    const float* __restrict__ src, bf16* __restrict__ dst, int R, int C)
{
  __shared__ float tile[32][33];
  const long long zoff = (long long)blockIdx.z * R * C;
  const int c0 = blockIdx.x * 32, r0 = blockIdx.y * 32;
  const int tx = (int)threadIdx.x & 31, ty = (int)threadIdx.x >> 5;
#pragma unroll
  for (int ii = 0; ii < 4; ++ii) {
    const int i = ty + ii * 8;
    if (r0 + i < R && c0 + tx < C)
      tile[i][tx] = src[zoff + (long long)(r0 + i) * C + (c0 + tx)];
  }
  __syncthreads();
#pragma unroll
  for (int ii = 0; ii < 4; ++ii) {
    const int i = ty + ii * 8;
    if (c0 + i < C && r0 + tx < R)
      dst[zoff + (long long)(c0 + i) * R + (r0 + tx)] = (bf16)tile[tx][i];
  }
}

// ---------------------------------------------------------------------------
// Row softmax (f32 -> bf16 probs), one 256-thread block per row
// ---------------------------------------------------------------------------
__global__ __launch_bounds__(256) void softmax_rows(
    const float* __restrict__ S, bf16* __restrict__ P, int cols, int ld)
{
  const long long row = blockIdx.x;
  const float* s = S + row * (long long)ld;
  bf16* p = P + row * (long long)ld;
  const int tid = (int)threadIdx.x;
  __shared__ float red[256];

  float mx = -3.402823466e38f;
  for (int c = tid; c < cols; c += 256) mx = fmaxf(mx, s[c]);
  red[tid] = mx; __syncthreads();
  for (int st = 128; st > 0; st >>= 1) { if (tid < st) red[tid] = fmaxf(red[tid], red[tid + st]); __syncthreads(); }
  mx = red[0]; __syncthreads();

  float sum = 0.f;
  for (int c = tid; c < cols; c += 256) sum += expf(s[c] - mx);
  red[tid] = sum; __syncthreads();
  for (int st = 128; st > 0; st >>= 1) { if (tid < st) red[tid] += red[tid + st]; __syncthreads(); }
  const float inv = 1.f / red[0];

  for (int c = tid; c < cols; c += 256) p[c] = (bf16)(expf(s[c] - mx) * inv);
}

__device__ __forceinline__ unsigned fkey(float f) {  // order-preserving f32->u32
  unsigned u = __float_as_uint(f);
  return (u & 0x80000000u) ? ~u : (u | 0x80000000u);
}

// ---------------------------------------------------------------------------
// Top-k thresholded softmax: keep s >= k-th largest (exact, 32-step binary
// search on monotone keys), one block per row.
// ---------------------------------------------------------------------------
__global__ __launch_bounds__(256) void sparse_softmax_rows(
    const float* __restrict__ S, bf16* __restrict__ P, int cols, int ld, int kth)
{
  const long long row = blockIdx.x;
  const float* s = S + row * (long long)ld;
  bf16* p = P + row * (long long)ld;
  const int tid = (int)threadIdx.x;
  __shared__ unsigned ured[256];
  __shared__ float    fred[256];

  unsigned lo = 0u, hi = 0xFFFFFFFFu;
  while (lo < hi) {
    const unsigned mid =
        (unsigned)(((unsigned long long)lo + (unsigned long long)hi + 1ull) >> 1);
    unsigned cnt = 0;
    for (int c = tid; c < cols; c += 256) cnt += (fkey(s[c]) >= mid) ? 1u : 0u;
    ured[tid] = cnt; __syncthreads();
    for (int st = 128; st > 0; st >>= 1) { if (tid < st) ured[tid] += ured[tid + st]; __syncthreads(); }
    cnt = ured[0]; __syncthreads();
    if (cnt >= (unsigned)kth) lo = mid; else hi = mid - 1u;
  }
  const unsigned thrK = lo;

  float mx = -3.402823466e38f;   // row max is always kept
  for (int c = tid; c < cols; c += 256) mx = fmaxf(mx, s[c]);
  fred[tid] = mx; __syncthreads();
  for (int st = 128; st > 0; st >>= 1) { if (tid < st) fred[tid] = fmaxf(fred[tid], fred[tid + st]); __syncthreads(); }
  mx = fred[0]; __syncthreads();

  float sum = 0.f;
  for (int c = tid; c < cols; c += 256)
    if (fkey(s[c]) >= thrK) sum += expf(s[c] - mx);
  fred[tid] = sum; __syncthreads();
  for (int st = 128; st > 0; st >>= 1) { if (tid < st) fred[tid] += fred[tid + st]; __syncthreads(); }
  const float inv = 1.f / fred[0];

  for (int c = tid; c < cols; c += 256) {
    const float v = (fkey(s[c]) >= thrK) ? expf(s[c] - mx) * inv : 0.f;
    p[c] = (bf16)v;
  }
}

// ---------------------------------------------------------------------------
// MoE router: rw = softmax(x2 @ Wr + br) over E=4, one block per row
// ---------------------------------------------------------------------------
__global__ __launch_bounds__(256) void router_softmax(
    const float* __restrict__ x2, const float* __restrict__ Wr,
    const float* __restrict__ br, float* __restrict__ rw, int D)
{
  const long long row = blockIdx.x;
  const float* xr = x2 + row * (long long)D;
  const int tid = (int)threadIdx.x;
  __shared__ float red[4][256];
  float a0 = 0.f, a1 = 0.f, a2 = 0.f, a3 = 0.f;
  for (int c = tid; c < D; c += 256) {
    const float xv = xr[c];
    const float* w = Wr + (long long)c * 4;
    a0 += xv * w[0]; a1 += xv * w[1]; a2 += xv * w[2]; a3 += xv * w[3];
  }
  red[0][tid] = a0; red[1][tid] = a1; red[2][tid] = a2; red[3][tid] = a3;
  __syncthreads();
  for (int st = 128; st > 0; st >>= 1) {
    if (tid < st) {
      red[0][tid] += red[0][tid + st]; red[1][tid] += red[1][tid + st];
      red[2][tid] += red[2][tid + st]; red[3][tid] += red[3][tid + st];
    }
    __syncthreads();
  }
  if (tid == 0) {
    float v0 = red[0][0] + br[0], v1 = red[1][0] + br[1];
    float v2 = red[2][0] + br[2], v3 = red[3][0] + br[3];
    const float m = fmaxf(fmaxf(v0, v1), fmaxf(v2, v3));
    const float e0 = expf(v0 - m), e1 = expf(v1 - m), e2 = expf(v2 - m), e3 = expf(v3 - m);
    const float inv = 1.f / (e0 + e1 + e2 + e3);
    rw[row * 4 + 0] = e0 * inv; rw[row * 4 + 1] = e1 * inv;
    rw[row * 4 + 2] = e2 * inv; rw[row * 4 + 3] = e3 * inv;
  }
}

// ---------------------------------------------------------------------------
// Elementwise helpers
// ---------------------------------------------------------------------------
__global__ void cvt_f32_bf16_kernel(const float* __restrict__ s, bf16* __restrict__ d, long long n) {
  const long long i = (long long)blockIdx.x * blockDim.x + threadIdx.x;
  if (i < n) d[i] = (bf16)s[i];
}
__global__ void residual_kernel(const float* __restrict__ x, const float* __restrict__ h,
                                float* __restrict__ x2f, bf16* __restrict__ x2b, long long n) {
  const long long i = (long long)blockIdx.x * blockDim.x + threadIdx.x;
  if (i < n) { const float v = x[i] + h[i]; x2f[i] = v; x2b[i] = (bf16)v; }
}
__global__ void copy_f32_kernel(const float* __restrict__ s, float* __restrict__ d, long long n) {
  const long long i = (long long)blockIdx.x * blockDim.x + threadIdx.x;
  if (i < n) d[i] = s[i];
}
__global__ void moe_acc_kernel(float* __restrict__ out, const float* __restrict__ eo,
                               const float* __restrict__ rw, int e, int D, long long n) {
  const long long i = (long long)blockIdx.x * blockDim.x + threadIdx.x;
  if (i < n) out[i] += rw[(i / (long long)D) * 4 + e] * eo[i];
}

// ---------------------------------------------------------------------------
// Host orchestration
// ---------------------------------------------------------------------------
extern "C" void kernel_launch(void* const* d_in, const int* in_sizes, int n_in,
                              void* d_out, int out_size, void* d_ws, size_t ws_size,
                              hipStream_t stream)
{
  (void)in_sizes; (void)n_in; (void)out_size; (void)ws_size;
  constexpr int B = 4, S = 2048, D = 1024, H = 256, E = 4, F = 4096, MMEM = 64;
  constexpr int MB = B * S;            // 8192 token rows
  constexpr int H3 = 3 * H;            // 768
  constexpr long long SS = (long long)S * S;
  int kth = (int)(S * 0.1); if (kth < 1) kth = 1;   // 204

  const float* x    = (const float*)d_in[0];
  const float* Wql  = (const float*)d_in[1];
  const float* Wqs  = (const float*)d_in[2];
  const float* memb = (const float*)d_in[3];
  const float* Wm   = (const float*)d_in[4];
  const float* Wo   = (const float*)d_in[5];
  const float* bo   = (const float*)d_in[6];
  const float* Wi   = (const float*)d_in[7];
  const float* bi   = (const float*)d_in[8];
  const float* Wp1  = (const float*)d_in[9];
  const float* bp1  = (const float*)d_in[10];
  const float* Wp2  = (const float*)d_in[11];
  const float* bp2  = (const float*)d_in[12];
  const float* Wwb  = (const float*)d_in[13];
  const float* bwb  = (const float*)d_in[14];
  const float* Wr   = (const float*)d_in[15];
  const float* br   = (const float*)d_in[16];
  const float* We1  = (const float*)d_in[17];
  const float* be1  = (const float*)d_in[18];
  const float* We2  = (const float*)d_in[19];
  const float* be2  = (const float*)d_in[20];
  float* out = (float*)d_out;

  char* base = (char*)d_ws;
  size_t off = 0;
  auto alloc = [&](size_t bytes) -> void* {
    void* p = base + off;
    off = (off + bytes + 255) & ~(size_t)255;
    return p;
  };

  // activations / intermediates
  bf16*  xbf   = (bf16*) alloc((size_t)MB * D * 2);
  bf16*  qkv   = (bf16*) alloc((size_t)MB * H3 * 2);      // [token][768]
  bf16*  qkvT  = (bf16*) alloc((size_t)H3 * MB * 2);      // [768][token]
  float* sc    = (float*)alloc((size_t)B * SS * 4);
  bf16*  pb    = (bf16*) alloc((size_t)B * SS * 2);
  bf16*  cat4  = (bf16*) alloc((size_t)MB * 4 * H * 2);
  bf16*  cat2  = (bf16*) alloc((size_t)MB * 2 * H * 2);
  float* mwS   = (float*)alloc((size_t)MB * MMEM * 4);
  bf16*  mwP   = (bf16*) alloc((size_t)MB * MMEM * 2);
  bf16*  ppb   = (bf16*) alloc((size_t)MB * H * 2);
  bf16*  t2b   = (bf16*) alloc((size_t)MB * 2 * H * 2);
  float* hyd   = (float*)alloc((size_t)MB * D * 4);
  float* x2f   = (float*)alloc((size_t)MB * D * 4);
  bf16*  x2b   = (bf16*) alloc((size_t)MB * D * 2);
  float* rw    = (float*)alloc((size_t)MB * E * 4);
  bf16*  hb    = (bf16*) alloc((size_t)MB * F * 2);
  float* eof   = (float*)alloc((size_t)MB * D * 4);
  // transposed bf16 weights (all GEMM B operands are N x K row-major)
  bf16*  WqlT  = (bf16*) alloc((size_t)H3 * D * 2);
  bf16*  WqsT  = (bf16*) alloc((size_t)H3 * D * 2);
  bf16*  WmT   = (bf16*) alloc((size_t)H * D * 2);
  bf16*  membf = (bf16*) alloc((size_t)MMEM * H * 2);     // plain cvt [64][256]
  bf16*  membT = (bf16*) alloc((size_t)H * MMEM * 2);     // [256][64]
  bf16*  WoT   = (bf16*) alloc((size_t)H * 2 * H * 2);
  bf16*  WiT   = (bf16*) alloc((size_t)H * D * 2);
  bf16*  Wp1T  = (bf16*) alloc((size_t)2 * H * H * 2);
  bf16*  Wp2T  = (bf16*) alloc((size_t)H * 2 * H * 2);
  bf16*  WwbT  = (bf16*) alloc((size_t)D * 4 * H * 2);
  bf16*  We1T  = (bf16*) alloc((size_t)E * F * D * 2);
  bf16*  We2T  = (bf16*) alloc((size_t)E * D * F * 2);

  auto cvt = [&](const float* s_, bf16* d_, long long n) {
    cvt_f32_bf16_kernel<<<dim3((unsigned)((n + 255) / 256)), 256, 0, stream>>>(s_, d_, n);
  };
  auto tcvt = [&](const float* s_, bf16* d_, int R, int C, int batch) {
    dim3 g((unsigned)((C + 31) / 32), (unsigned)((R + 31) / 32), (unsigned)batch);
    transpose_cvt_kernel<<<g, 256, 0, stream>>>(s_, d_, R, C);
  };
  auto gemm = [&](const bf16* A, long long sA, int lda,
                  const bf16* Bp, long long sB, int ldb,
                  float* Cf, bf16* Cb, bf16* CbT,
                  long long sC, int ldc, int ldct, long long sCT,
                  const float* bias, float alpha, int act,
                  int M, int N, int K, int batch) {
    dim3 grid((unsigned)((N + TN - 1) / TN), (unsigned)((M + TM - 1) / TM), (unsigned)batch);
    gemm_bt_wmma<<<grid, 256, 0, stream>>>(A, sA, lda, Bp, sB, ldb,
                                           Cf, Cb, CbT, sC, ldc, ldct, sCT,
                                           bias, alpha, act, M, N, K);
  };

  // ---- convert activations + transpose-convert all weights (L2-resident) ----
  cvt(x, xbf, (long long)MB * D);
  tcvt(Wql,  WqlT, D, H3, 1);
  tcvt(Wqs,  WqsT, D, H3, 1);
  tcvt(Wm,   WmT,  D, H, 1);
  cvt(memb,  membf, (long long)MMEM * H);
  tcvt(memb, membT, MMEM, H, 1);
  tcvt(Wo,   WoT,  2 * H, H, 1);
  tcvt(Wi,   WiT,  D, H, 1);
  tcvt(Wp1,  Wp1T, H, 2 * H, 1);
  tcvt(Wp2,  Wp2T, 2 * H, H, 1);
  tcvt(Wwb,  WwbT, 4 * H, D, 1);
  tcvt(We1,  We1T, D, F, E);
  tcvt(We2,  We2T, F, D, E);

  const float scaleH = 0.0625f;   // 1/sqrt(256)

  // ---- local attention -> cat4[:, 0:H] ----
  gemm(xbf, 0, D, WqlT, 0, D, nullptr, qkv, qkvT, 0, H3, MB, 0,
       nullptr, 1.f, 0, MB, H3, D, 1);
  gemm(qkv, (long long)S * H3, H3, qkv + H, (long long)S * H3, H3,
       sc, nullptr, nullptr, SS, S, 0, 0, nullptr, scaleH, 0, S, S, H, B);
  softmax_rows<<<dim3((unsigned)MB), 256, 0, stream>>>(sc, pb, S, S);
  gemm(pb, SS, S, qkvT + (long long)2 * H * MB, (long long)S, MB,
       nullptr, cat4, nullptr, (long long)S * 4 * H, 4 * H, 0, 0,
       nullptr, 1.f, 0, S, H, S, B);

  // ---- sparse attention -> cat4[:, H:2H] (reuses qkv/qkvT/sc/pb) ----
  gemm(xbf, 0, D, WqsT, 0, D, nullptr, qkv, qkvT, 0, H3, MB, 0,
       nullptr, 1.f, 0, MB, H3, D, 1);
  gemm(qkv, (long long)S * H3, H3, qkv + H, (long long)S * H3, H3,
       sc, nullptr, nullptr, SS, S, 0, 0, nullptr, scaleH, 0, S, S, H, B);
  sparse_softmax_rows<<<dim3((unsigned)MB), 256, 0, stream>>>(sc, pb, S, S, kth);
  gemm(pb, SS, S, qkvT + (long long)2 * H * MB, (long long)S, MB,
       nullptr, cat4 + H, nullptr, (long long)S * 4 * H, 4 * H, 0, 0,
       nullptr, 1.f, 0, S, H, S, B);

  // ---- memory path -> cat4[:, 2H:3H] ----
  gemm(xbf, 0, D, WmT, 0, D, nullptr, cat2, nullptr, 0, 2 * H, 0, 0,
       nullptr, 1.f, 0, MB, H, D, 1);
  gemm(cat2, 0, 2 * H, membf, 0, H, mwS, nullptr, nullptr, 0, MMEM, 0, 0,
       nullptr, 1.f, 0, MB, MMEM, H, 1);
  softmax_rows<<<dim3((unsigned)MB), 256, 0, stream>>>(mwS, mwP, MMEM, MMEM);
  gemm(mwP, 0, MMEM, membT, 0, MMEM, nullptr, cat2 + H, nullptr, 0, 2 * H, 0, 0,
       nullptr, 1.f, 0, MB, H, MMEM, 1);
  gemm(cat2, 0, 2 * H, WoT, 0, 2 * H, nullptr, cat4 + 2 * H, nullptr, 0, 4 * H, 0, 0,
       bo, 1.f, 0, MB, H, 2 * H, 1);

  // ---- predictor path -> cat4[:, 3H:4H] ----
  gemm(xbf, 0, D, WiT, 0, D, nullptr, ppb, nullptr, 0, H, 0, 0,
       bi, 1.f, 0, MB, H, D, 1);
  gemm(ppb, 0, H, Wp1T, 0, H, nullptr, t2b, nullptr, 0, 2 * H, 0, 0,
       bp1, 1.f, 1, MB, 2 * H, H, 1);
  gemm(t2b, 0, 2 * H, Wp2T, 0, 2 * H, nullptr, cat4 + 3 * H, nullptr, 0, 4 * H, 0, 0,
       bp2, 1.f, 0, MB, H, 2 * H, 1);

  // ---- hydra merge + residual ----
  gemm(cat4, 0, 4 * H, WwbT, 0, 4 * H, hyd, nullptr, nullptr, 0, D, 0, 0,
       bwb, 1.f, 0, MB, D, 4 * H, 1);
  {
    const long long n = (long long)MB * D;
    residual_kernel<<<dim3((unsigned)((n + 255) / 256)), 256, 0, stream>>>(x, hyd, x2f, x2b, n);
  }

  // ---- MoE: router + dense expert evaluation (dominant FLOPs, all WMMA) ----
  router_softmax<<<dim3((unsigned)MB), 256, 0, stream>>>(x2f, Wr, br, rw, D);
  {
    const long long n = (long long)MB * D;
    copy_f32_kernel<<<dim3((unsigned)((n + 255) / 256)), 256, 0, stream>>>(x2f, out, n);
  }
  for (int e = 0; e < E; ++e) {
    gemm(x2b, 0, D, We1T + (long long)e * F * D, 0, D,
         nullptr, hb, nullptr, 0, F, 0, 0,
         be1 + (long long)e * F, 1.f, 1, MB, F, D, 1);
    gemm(hb, 0, F, We2T + (long long)e * D * F, 0, F,
         eof, nullptr, nullptr, 0, D, 0, 0,
         be2 + (long long)e * D, 1.f, 0, MB, D, F, 1);
    const long long n = (long long)MB * D;
    moe_acc_kernel<<<dim3((unsigned)((n + 255) / 256)), 256, 0, stream>>>(out, eof, rw, e, D, n);
  }
}
